// FlashAttention_77309411706
// MI455X (gfx1250) — matrix operand order repaired and checked by reading, not compiled, not run on hardware
//
#include <hip/hip_runtime.h>

// ---------------------------------------------------------------------------
// Fused QKV-projection + RoPE + FlashAttention for MI455X (gfx1250, wave32)
//   hidden (4,2048,1024) f32 ; wqkv (3072,1024) f32 ; bqkv (3072) f32
//   out    (4,2048,1024) f32
// All GEMMs on v_wmma_f32_16x16x32_f16 (f16 in, f32 accumulate).
// Pipeline:
//   k0: one-pass f32->f16 conversion of H and W into workspace (amortizes
//       the cvt that was previously repeated per wave/tile in the GEMM loop)
//   k1: QKV GEMM (WMMA) + bias + RoPE, writes Q/K [b,h,s,d] f16 and
//       V transposed [b,h,d,s] f16
//   k2: flash attention (WMMA QK^T and PV, online softmax), f32 out
// ---------------------------------------------------------------------------

typedef __attribute__((ext_vector_type(16))) _Float16 v16h;
typedef __attribute__((ext_vector_type(8)))  _Float16 v8h;
typedef __attribute__((ext_vector_type(8)))  float    v8f;
typedef __attribute__((ext_vector_type(4)))  float    v4f;

#define BATCH   4
#define SEQ     2048
#define HDIM    1024
#define NHEADS  16
#define DHEAD   64
#define NQKV    3072   // 3*HDIM

static __device__ __forceinline__ v8f wmma_f16(const v16h& a, const v16h& b, const v8f& c) {
    return __builtin_amdgcn_wmma_f32_16x16x32_f16(
        /*neg_a=*/false, a, /*neg_b=*/false, b,
        /*c_mod=*/(short)0, c, /*reuse_a=*/false, /*reuse_b=*/false);
}

// ===========================================================================
// Kernel 0: f32 -> f16 conversion, vectorized 8 floats / thread / step.
// ===========================================================================
__global__ __launch_bounds__(256) void cvt_f32_to_f16_kernel(
    const float* __restrict__ src, _Float16* __restrict__ dst, int n8)
{
    int i = blockIdx.x * blockDim.x + threadIdx.x;
    int stride = gridDim.x * blockDim.x;
    for (; i < n8; i += stride) {
        const float* p = src + (size_t)i * 8;
        v4f f0 = *(const v4f*)(p);
        v4f f1 = *(const v4f*)(p + 4);
        v8h h;
        h[0]=(_Float16)f0.x; h[1]=(_Float16)f0.y; h[2]=(_Float16)f0.z; h[3]=(_Float16)f0.w;
        h[4]=(_Float16)f1.x; h[5]=(_Float16)f1.y; h[6]=(_Float16)f1.z; h[7]=(_Float16)f1.w;
        *(v8h*)(dst + (size_t)i * 8) = h;
    }
}

// ===========================================================================
// Kernel 1: C[t, n] = H[t, :] . W[n, :] + bias[n], t = b*2048+s, n in [0,3072)
// Block: 256 thr (8 waves). Block tile = 128 rows x 64 cols.
// 64-col tile == one (qkv, head) pair, so RoPE pairing (dd, dd+32) maps to
// accumulators (acc[j], acc[j+2]) lane-for-lane.
// Inner loop is pure b128 f16 loads + WMMA (no converts).
// Outputs (f16):  Qh,Kh: [b][h][s][d]    Vt: [b][h][d][s]  (transposed)
// ===========================================================================
__global__ __launch_bounds__(256) void qkv_rope_kernel(
    const _Float16* __restrict__ Hf, const _Float16* __restrict__ Wf,
    const float* __restrict__ bias,
    _Float16* __restrict__ Qh, _Float16* __restrict__ Kh,
    _Float16* __restrict__ Vt)
{
    const int lane   = threadIdx.x & 31;
    const int wave   = threadIdx.x >> 5;
    const int half16 = lane >> 4;      // which 16-lane group
    const int col    = lane & 15;

    const int t0   = blockIdx.x * 128 + wave * 16;  // first token row of wave
    const int n0   = blockIdx.y * 64;               // first output column
    const int qkv  = n0 >> 10;                      // 0=q 1=k 2=v
    const int head = (n0 & 1023) >> 6;

    const _Float16* Arow = Hf + (size_t)(t0 + col) * HDIM;  // A-layout: row = lane&15
    const _Float16* Brow[4];
#pragma unroll
    for (int j = 0; j < 4; ++j)
        Brow[j] = Wf + (size_t)(n0 + j * 16 + col) * HDIM;  // B-layout: col = lane&15

    v8f zero = {};
    v8f acc[4];
#pragma unroll
    for (int j = 0; j < 4; ++j) acc[j] = zero;

    for (int k0 = 0; k0 < HDIM; k0 += 32) {
        // prefetch next k-step of A (lowers to global_prefetch_b8)
        if (k0 + 32 < HDIM) __builtin_prefetch(Arow + k0 + 32, 0, 1);

        // ---- A fragment (16x32 f16): elems 0..7  -> K = half16*8 + 0..7,
        //                              elems 8..15 -> K = 16 + half16*8 + 0..7
        v16h a;
        {
            v8h lo = *(const v8h*)(Arow + k0 + half16 * 8);
            v8h hi = *(const v8h*)(Arow + k0 + 16 + half16 * 8);
#pragma unroll
            for (int i = 0; i < 8; ++i) { a[i] = lo[i]; a[8 + i] = hi[i]; }
        }
        // ---- B fragments (32x16 f16): same K order as A (half16*8 + e, 16 + half16*8 + e)
#pragma unroll
        for (int j = 0; j < 4; ++j) {
            v8h lo = *(const v8h*)(Brow[j] + k0 + half16 * 8);
            v8h hi = *(const v8h*)(Brow[j] + k0 + 16 + half16 * 8);
            v16h bf;
#pragma unroll
            for (int i = 0; i < 8; ++i) { bf[i] = lo[i]; bf[8 + i] = hi[i]; }
            acc[j] = wmma_f16(a, bf, acc[j]);
        }
    }

    // ---- bias (C-layout: col of acc[j] = n0 + j*16 + (lane&15), all 8 rows)
#pragma unroll
    for (int j = 0; j < 4; ++j) {
        float bj = bias[n0 + j * 16 + col];
#pragma unroll
        for (int r = 0; r < 8; ++r) acc[j][r] += bj;
    }

    const size_t bh = (size_t)((t0 >> 11) * NHEADS + head);

    if (qkv == 2) {
        // V: store transposed f16  Vt[bh][d][s]
#pragma unroll
        for (int j = 0; j < 4; ++j) {
            _Float16* dst = Vt + (bh * DHEAD + (j * 16 + col)) * SEQ;
#pragma unroll
            for (int r = 0; r < 8; ++r) {
                int s = (t0 + r + 8 * half16) & (SEQ - 1);
                dst[s] = (_Float16)acc[j][r];
            }
        }
    } else {
        // Q/K: apply RoPE, store f16 [bh][s][d]
        _Float16* dstQK = (qkv == 0 ? Qh : Kh) + bh * SEQ * DHEAD;
        float invf[2];
        invf[0] = __powf(10000.0f, -(float)(col)      * (1.0f / 32.0f));
        invf[1] = __powf(10000.0f, -(float)(16 + col) * (1.0f / 32.0f));
#pragma unroll
        for (int j = 0; j < 2; ++j) {
            int dd = j * 16 + col;              // dd < 32, pairs with dd+32
#pragma unroll
            for (int r = 0; r < 8; ++r) {
                int s = (t0 + r + 8 * half16) & (SEQ - 1);
                float sv, cv;
                __sincosf((float)s * invf[j], &sv, &cv);
                float x1 = acc[j][r];          // element dd
                float x2 = acc[j + 2][r];      // element dd+32 (same lane!)
                _Float16* drow = dstQK + (size_t)s * DHEAD;
                drow[dd]      = (_Float16)(x1 * cv - x2 * sv);
                drow[dd + 32] = (_Float16)(x2 * cv + x1 * sv);
            }
        }
    }
}

// ===========================================================================
// Kernel 2: flash attention per (b, head). Block: 256 thr = 8 waves,
// each wave owns 16 query rows (block covers 128). Key blocks of 32.
//   S = (Q*K^T)/sqrt(d)  -> online softmax -> O += P*V, O /= l
// Q fragment A-layout loads are contiguous f16 b128; K is [s][d] so the
// B-fragment (N=key row, K=d) is two contiguous 8-half (b128) reads; V was stored
// transposed [d][s] so the P*V B-fragment (N=d col, K=key) is contiguous too.
// P (C-layout) -> A-layout via a wave-private LDS bounce.
// ===========================================================================
__global__ __launch_bounds__(256) void flash_attn_kernel(
    const _Float16* __restrict__ Qh, const _Float16* __restrict__ Kh,
    const _Float16* __restrict__ Vt, float* __restrict__ out)
{
    // 8 waves * 16 rows * 48-half row stride (96B, keeps b128 reads aligned)
    __shared__ _Float16 psmem[8 * 16 * 48];

    const int lane   = threadIdx.x & 31;
    const int wave   = threadIdx.x >> 5;
    const int half16 = lane >> 4;
    const int col    = lane & 15;

    const int head = blockIdx.y;
    const int b    = blockIdx.z;
    const size_t bh = (size_t)(b * NHEADS + head);
    const int q0 = blockIdx.x * 128 + wave * 16;

    // ---- Q fragments (pre-scaled by 1/sqrt(64) = 0.125), kept for whole loop
    const _Float16* Qrow = Qh + (bh * SEQ + (q0 + col)) * DHEAD;
    v16h qa[2];
#pragma unroll
    for (int ds = 0; ds < 2; ++ds) {
        v8h lo = *(const v8h*)(Qrow + ds * 32 + half16 * 8);
        v8h hi = *(const v8h*)(Qrow + ds * 32 + 16 + half16 * 8);
#pragma unroll
        for (int i = 0; i < 8; ++i) {
            qa[ds][i]     = lo[i] * (_Float16)0.125f;
            qa[ds][8 + i] = hi[i] * (_Float16)0.125f;
        }
    }

    v8f zero = {};
    v8f o[4];
#pragma unroll
    for (int jd = 0; jd < 4; ++jd) o[jd] = zero;
    float mrow[8], lsum[8];
#pragma unroll
    for (int r = 0; r < 8; ++r) { mrow[r] = -1e30f; lsum[r] = 0.0f; }

    _Float16* pbuf = psmem + wave * (16 * 48);   // wave-private

    for (int kb = 0; kb < SEQ; kb += 32) {
        // ---- scores: S(16x32) = Qtile * K^T, two 16x16 n-subtiles
        v8f s0 = zero, s1 = zero;
#pragma unroll
        for (int j = 0; j < 2; ++j) {
            const _Float16* Krow = Kh + (bh * SEQ + (kb + j * 16 + col)) * DHEAD;
#pragma unroll
            for (int ds = 0; ds < 2; ++ds) {
                v8h klo = *(const v8h*)(Krow + ds * 32 + half16 * 8);
                v8h khi = *(const v8h*)(Krow + ds * 32 + 16 + half16 * 8);
                v16h kf;
#pragma unroll
                for (int i = 0; i < 8; ++i) { kf[i] = klo[i]; kf[8 + i] = khi[i]; }
                if (j == 0) s0 = wmma_f16(qa[ds], kf, s0);
                else        s1 = wmma_f16(qa[ds], kf, s1);
            }
        }

        // ---- online softmax over this wave's 16 rows
        float a_fac[8];
#pragma unroll
        for (int r = 0; r < 8; ++r) {
            float mx = fmaxf(s0[r], s1[r]);
            mx = fmaxf(mx, __shfl_xor(mx, 1, 32));
            mx = fmaxf(mx, __shfl_xor(mx, 2, 32));
            mx = fmaxf(mx, __shfl_xor(mx, 4, 32));
            mx = fmaxf(mx, __shfl_xor(mx, 8, 32));
            float mnew  = fmaxf(mrow[r], mx);
            float alpha = __expf(mrow[r] - mnew);
            mrow[r] = mnew;
            float p0 = __expf(s0[r] - mnew);
            float p1 = __expf(s1[r] - mnew);
            s0[r] = p0; s1[r] = p1;
            float rs = p0 + p1;
            rs += __shfl_xor(rs, 1, 32);
            rs += __shfl_xor(rs, 2, 32);
            rs += __shfl_xor(rs, 4, 32);
            rs += __shfl_xor(rs, 8, 32);
            lsum[r] = lsum[r] * alpha + rs;
            a_fac[r] = alpha;
        }
#pragma unroll
        for (int jd = 0; jd < 4; ++jd)
#pragma unroll
            for (int r = 0; r < 8; ++r) o[jd][r] *= a_fac[r];

        // ---- P: C-layout -> A-layout via LDS (wave-private region)
#pragma unroll
        for (int r = 0; r < 8; ++r) {
            int m = r + 8 * half16;
            pbuf[m * 48 + col]      = (_Float16)s0[r];
            pbuf[m * 48 + 16 + col] = (_Float16)s1[r];
        }
        asm volatile("s_wait_dscnt 0" ::: "memory");
        v16h pf;
        {
            const _Float16* prow = pbuf + col * 48 + half16 * 8;
            v8h lo = *(const v8h*)(prow);
            v8h hi = *(const v8h*)(prow + 16);
#pragma unroll
            for (int i = 0; i < 8; ++i) { pf[i] = lo[i]; pf[8 + i] = hi[i]; }
        }

        // ---- O += P * Vblk : V transposed, two contiguous 8-key reads per lane
#pragma unroll
        for (int jd = 0; jd < 4; ++jd) {
            const _Float16* Vrow = Vt + (bh * DHEAD + (jd * 16 + col)) * SEQ
                                      + kb + half16 * 8;
            v8h vlo = *(const v8h*)(Vrow);
            v8h vhi = *(const v8h*)(Vrow + 16);
            v16h vf;
#pragma unroll
            for (int i = 0; i < 8; ++i) { vf[i] = vlo[i]; vf[8 + i] = vhi[i]; }
            o[jd] = wmma_f16(pf, vf, o[jd]);
        }
    }

    // ---- finalize: O /= l, write f32 out[b][s][head*64 + d]
#pragma unroll
    for (int r = 0; r < 8; ++r) {
        float inv = 1.0f / lsum[r];
        int m = r + 8 * half16;
        float* dst = out + ((size_t)b * SEQ + q0 + m) * HDIM + head * DHEAD;
#pragma unroll
        for (int jd = 0; jd < 4; ++jd)
            dst[jd * 16 + col] = o[jd][r] * inv;
    }
}

// ===========================================================================
extern "C" void kernel_launch(void* const* d_in, const int* in_sizes, int n_in,
                              void* d_out, int out_size, void* d_ws, size_t ws_size,
                              hipStream_t stream) {
    (void)in_sizes; (void)n_in; (void)out_size; (void)ws_size;
    const float* H    = (const float*)d_in[0];   // (4,2048,1024)
    const float* W    = (const float*)d_in[1];   // (3072,1024)
    const float* bias = (const float*)d_in[2];   // (3072,)
    float* out = (float*)d_out;

    const size_t nH = (size_t)BATCH * SEQ * HDIM;            // 8,388,608
    const size_t nW = (size_t)NQKV * HDIM;                   // 3,145,728
    const size_t perTensor = (size_t)BATCH * NHEADS * SEQ * DHEAD;  // 8M f16

    _Float16* Qh = (_Float16*)d_ws;
    _Float16* Kh = Qh + perTensor;
    _Float16* Vt = Kh + perTensor;
    _Float16* Hf = Vt + perTensor;
    _Float16* Wf = Hf + nH;                                  // ~74 MB total

    // k0: one-pass f32 -> f16 conversion (bandwidth-bound, ~2 us)
    cvt_f32_to_f16_kernel<<<1024, 256, 0, stream>>>(H, Hf, (int)(nH / 8));
    cvt_f32_to_f16_kernel<<<512, 256, 0, stream>>>(W, Wf, (int)(nW / 8));

    // k1: QKV GEMM + bias + RoPE
    dim3 g1((BATCH * SEQ) / 128, NQKV / 64);     // 64 x 48 blocks
    qkv_rope_kernel<<<g1, 256, 0, stream>>>(Hf, Wf, bias, Qh, Kh, Vt);

    // k2: flash attention
    dim3 g2(SEQ / 128, NHEADS, BATCH);           // 16 x 16 x 4 blocks
    flash_attn_kernel<<<g2, 256, 0, stream>>>(Qh, Kh, Vt, out);
}
